// StaticCombine_24524263260160
// MI455X (gfx1250) — matrix operand order repaired and checked by reading, not compiled, not run on hardware
//
#include <hip/hip_runtime.h>

typedef unsigned int u32;
typedef __attribute__((ext_vector_type(16))) __bf16 v16bf;
typedef __attribute__((ext_vector_type(8)))  float  v8f;

#define FDIM 1024
#define HDIM 64
#define NVAR 16
#define EPSF 1e-5f

// fragment table (units of 256 dwords = one 32x16 bf16 B-fragment)
#define FR_W1F 0      // 32 kc x 4 nt = 128
#define FR_WSF 128    // 32 kc x 1 nt = 32
#define FR_W2F 160    // 2 kc x 4 nt = 8
#define FR_WVF 168    // 2
#define FR_WGF 170    // 2
#define FR_W1V 172    // 16 vars x 8
#define FR_W2V 300
#define FR_WVV 428
#define FR_WGV 556
#define FR_TOTAL 684  // 684 KB of d_ws needed

__device__ __forceinline__ u32 pack_bf2(float a, float b) {
  __bf16 t[2] = {(__bf16)a, (__bf16)b};     // native RNE cvt -> v_cvt_pk_bf16_f32
  u32 d; __builtin_memcpy(&d, t, 4); return d;
}

__device__ __forceinline__ v8f wmma_bf16(v16bf a, v16bf b, v8f c) {
  return __builtin_amdgcn_wmma_f32_16x16x32_bf16(false, a, false, b, (short)0, c,
                                                 false, false);
}

// B fragment: packed buffer, lane-major (lane*8 dwords), two b128 loads
__device__ __forceinline__ v16bf load_bfrag(const u32* __restrict__ pk, int frag, int lane) {
  const uint4* p = (const uint4*)(pk + ((size_t)frag << 8) + lane * 8);
  uint4 q0 = p[0], q1 = p[1];
  u32 u[8] = {q0.x, q0.y, q0.z, q0.w, q1.x, q1.y, q1.z, q1.w};
  v16bf r; __builtin_memcpy(&r, u, 32); return r;
}

// A fragment (16x32 bf16) from f32 row-major global memory.
// lane<16: row=lane, K {0..7,16..23}; lane>=16: row=lane-16, K {8..15,24..31}
__device__ __forceinline__ v16bf load_afrag_g(const float* __restrict__ rowp, int cb, int kh) {
  const float* s = rowp + cb + kh * 8;
  float4 f0 = *(const float4*)(s);
  float4 f1 = *(const float4*)(s + 4);
  float4 f2 = *(const float4*)(s + 16);
  float4 f3 = *(const float4*)(s + 20);
  v16bf r;
  r[0]  = (__bf16)f0.x; r[1]  = (__bf16)f0.y; r[2]  = (__bf16)f0.z; r[3]  = (__bf16)f0.w;
  r[4]  = (__bf16)f1.x; r[5]  = (__bf16)f1.y; r[6]  = (__bf16)f1.z; r[7]  = (__bf16)f1.w;
  r[8]  = (__bf16)f2.x; r[9]  = (__bf16)f2.y; r[10] = (__bf16)f2.z; r[11] = (__bf16)f2.w;
  r[12] = (__bf16)f3.x; r[13] = (__bf16)f3.y; r[14] = (__bf16)f3.z; r[15] = (__bf16)f3.w;
  return r;
}

// A fragment from a 16x64 bf16 LDS tile (ds_load_b128 x2)
__device__ __forceinline__ v16bf load_afrag_lds(const __bf16* hl, int c, int lo, int kh) {
  const __bf16* s = hl + lo * 64 + c * 32 + kh * 8;
  uint4 q0 = *(const uint4*)(s);
  uint4 q1 = *(const uint4*)(s + 16);
  u32 u[8] = {q0.x, q0.y, q0.z, q0.w, q1.x, q1.y, q1.z, q1.w};
  v16bf r; __builtin_memcpy(&r, u, 32); return r;
}

__device__ __forceinline__ float redsum16(float v) {
  v += __shfl_xor(v, 1, 32);
  v += __shfl_xor(v, 2, 32);
  v += __shfl_xor(v, 4, 32);
  v += __shfl_xor(v, 8, 32);
  return v;
}
__device__ __forceinline__ float redmax16(float v) {
  v = fmaxf(v, __shfl_xor(v, 1, 32));
  v = fmaxf(v, __shfl_xor(v, 2, 32));
  v = fmaxf(v, __shfl_xor(v, 4, 32));
  v = fmaxf(v, __shfl_xor(v, 8, 32));
  return v;
}

// ---------------------------------------------------------------------------
// Kernel 1: repack all weight matrices (f32 row-major [K,Nout]) into bf16
// WMMA B-fragments. One block per fragment, one thread per packed dword.
// ---------------------------------------------------------------------------
__global__ void pack_weights(const float* __restrict__ W1f, const float* __restrict__ W2f,
                             const float* __restrict__ Wvf, const float* __restrict__ Wgf,
                             const float* __restrict__ Wsf, const float* __restrict__ W1v,
                             const float* __restrict__ W2v, const float* __restrict__ Wvv,
                             const float* __restrict__ Wgv, u32* __restrict__ pk) {
  int frag = blockIdx.x;
  int t = threadIdx.x;          // 0..255
  int lane = t >> 3, v = t & 7;
  int nl = lane & 15, kh = lane >> 4;
  int kl = (v >> 2) * 16 + kh * 8 + 2 * (v & 3);     // K within 32-chunk for this dword

  const float* W; int Nout, kc, nt;
  if (frag < FR_WSF)           { W = W1f; Nout = 64; int f = frag;          kc = f >> 2; nt = f & 3; }
  else if (frag < FR_W2F)      { W = Wsf; Nout = 16; kc = frag - FR_WSF;    nt = 0; }
  else if (frag < FR_WVF)      { W = W2f; Nout = 64; int f = frag - FR_W2F; kc = f >> 2; nt = f & 3; }
  else if (frag < FR_WGF)      { W = Wvf; Nout = 16; kc = frag - FR_WVF;    nt = 0; }
  else if (frag < FR_W1V)      { W = Wgf; Nout = 16; kc = frag - FR_WGF;    nt = 0; }
  else {
    int f = frag - FR_W1V;             // 0..511
    int grp = f >> 7;                  // which per-variable matrix
    int r = f & 127;
    int n = r >> 3, sub = r & 7;
    const float* base = (grp == 0) ? W1v : (grp == 1) ? W2v : (grp == 2) ? Wvv : Wgv;
    W = base + (size_t)n * 64 * 64; Nout = 64; kc = sub >> 2; nt = sub & 3;
  }
  int k = kc * 32 + kl;
  int n = nt * 16 + nl;
  float a = W[(size_t)k * Nout + n];
  float b = W[(size_t)(k + 1) * Nout + n];
  pk[((size_t)frag << 8) + lane * 8 + v] = pack_bf2(a, b);
}

// ---------------------------------------------------------------------------
// Kernel 2: fused VSN. 8 waves/block, each wave owns a 16-row batch tile.
// ---------------------------------------------------------------------------
__global__ __launch_bounds__(256) void vsn_fused(
    const float* __restrict__ emb,
    const float* __restrict__ b1f, const float* __restrict__ b2f,
    const float* __restrict__ bvf, const float* __restrict__ bgf,
    const float* __restrict__ bsf, const float* __restrict__ gf,
    const float* __restrict__ betaf,
    const float* __restrict__ b1v, const float* __restrict__ b2v,
    const float* __restrict__ bvv, const float* __restrict__ bgv,
    const float* __restrict__ gv, const float* __restrict__ betav,
    const u32* __restrict__ pk,
    float* __restrict__ out_static, float* __restrict__ out_sw, int Btot) {
  __shared__ __align__(16) __bf16 smem[8 * 16 * 64];  // 16 KB: bf16 h-tile per wave
  const int lane = threadIdx.x & 31;
  const int wave = threadIdx.x >> 5;
  const int lo = lane & 15;
  const int kh = lane >> 4;
  __bf16* hl = smem + wave * (16 * 64);
  const int row0 = (blockIdx.x * 8 + wave) * 16;
  if (row0 >= Btot) return;
  const float* erow = emb + (size_t)(row0 + lo) * FDIM;  // this lane's A row

  // ---- Phase 1: flattened GRN layer 1 + skip projection (K = 1024) ----
  v8f acc[4] = {};
  v8f accs = {};
  for (int kc = 0; kc < 32; ++kc) {
    v16bf a = load_afrag_g(erow, kc * 32, kh);
    if (kc < 31) __builtin_prefetch(erow + (kc + 1) * 32 + kh * 8, 0, 0);
    acc[0] = wmma_bf16(a, load_bfrag(pk, FR_W1F + kc * 4 + 0, lane), acc[0]);
    acc[1] = wmma_bf16(a, load_bfrag(pk, FR_W1F + kc * 4 + 1, lane), acc[1]);
    acc[2] = wmma_bf16(a, load_bfrag(pk, FR_W1F + kc * 4 + 2, lane), acc[2]);
    acc[3] = wmma_bf16(a, load_bfrag(pk, FR_W1F + kc * 4 + 3, lane), acc[3]);
    accs   = wmma_bf16(a, load_bfrag(pk, FR_WSF + kc, lane), accs);
  }
  // bias + ELU -> LDS (D layout -> row-major bf16 tile)
  #pragma unroll
  for (int nt = 0; nt < 4; ++nt) {
    float bb = b1f[nt * 16 + lo];
    #pragma unroll
    for (int r = 0; r < 8; ++r) {
      float x = acc[nt][r] + bb;
      x = x > 0.f ? x : (__expf(x) - 1.f);
      hl[(r + 8 * kh) * 64 + nt * 16 + lo] = (__bf16)x;
    }
  }
  v16bf h0 = load_afrag_lds(hl, 0, lo, kh);
  v16bf h1 = load_afrag_lds(hl, 1, lo, kh);

  // ---- layer 2 (64x64) ----
  v8f a2[4] = {};
  #pragma unroll
  for (int nt = 0; nt < 4; ++nt) {
    a2[nt] = wmma_bf16(h0, load_bfrag(pk, FR_W2F + nt, lane), a2[nt]);
    a2[nt] = wmma_bf16(h1, load_bfrag(pk, FR_W2F + 4 + nt, lane), a2[nt]);
  }
  #pragma unroll
  for (int nt = 0; nt < 4; ++nt) {
    float bb = b2f[nt * 16 + lo];
    #pragma unroll
    for (int r = 0; r < 8; ++r)
      hl[(r + 8 * kh) * 64 + nt * 16 + lo] = (__bf16)(a2[nt][r] + bb);
  }
  h0 = load_afrag_lds(hl, 0, lo, kh);
  h1 = load_afrag_lds(hl, 1, lo, kh);

  // ---- GLU + skip + LayerNorm(16) + softmax -> sparse weights ----
  v8f av = {}, ag = {};
  av = wmma_bf16(h0, load_bfrag(pk, FR_WVF + 0, lane), av);
  av = wmma_bf16(h1, load_bfrag(pk, FR_WVF + 1, lane), av);
  ag = wmma_bf16(h0, load_bfrag(pk, FR_WGF + 0, lane), ag);
  ag = wmma_bf16(h1, load_bfrag(pk, FR_WGF + 1, lane), ag);

  float sw[8];
  {
    float bv_ = bvf[lo], bg_ = bgf[lo], bs_ = bsf[lo];
    float g_ = gf[lo], be_ = betaf[lo];
    #pragma unroll
    for (int r = 0; r < 8; ++r) {
      float vv = av[r] + bv_;
      float gg = ag[r] + bg_;
      float gl = vv / (1.f + __expf(-gg));
      float y = accs[r] + bs_ + gl;
      float s = redsum16(y);
      float q = redsum16(y * y);
      float mu = s * (1.f / 16.f);
      float var = q * (1.f / 16.f) - mu * mu;
      float lg = g_ * (y - mu) * rsqrtf(var + EPSF) + be_;
      float mx = redmax16(lg);
      float e = __expf(lg - mx);
      float se = redsum16(e);
      sw[r] = e / se;
      out_sw[(size_t)(row0 + r + 8 * kh) * 16 + lo] = sw[r];
    }
  }

  // ---- Phase 4: 16 per-variable GRNs + LN(64) + weighted sum ----
  v8f sv[4] = {};
  for (int n = 0; n < NVAR; ++n) {
    v16bf a0 = load_afrag_g(erow, n * 64, kh);
    v16bf a1 = load_afrag_g(erow, n * 64 + 32, kh);
    v8f t1[4] = {};
    #pragma unroll
    for (int nt = 0; nt < 4; ++nt) {
      t1[nt] = wmma_bf16(a0, load_bfrag(pk, FR_W1V + n * 8 + nt, lane), t1[nt]);
      t1[nt] = wmma_bf16(a1, load_bfrag(pk, FR_W1V + n * 8 + 4 + nt, lane), t1[nt]);
    }
    #pragma unroll
    for (int nt = 0; nt < 4; ++nt) {
      float bb = b1v[n * 64 + nt * 16 + lo];
      #pragma unroll
      for (int r = 0; r < 8; ++r) {
        float x = t1[nt][r] + bb;
        x = x > 0.f ? x : (__expf(x) - 1.f);
        hl[(r + 8 * kh) * 64 + nt * 16 + lo] = (__bf16)x;
      }
    }
    v16bf g0 = load_afrag_lds(hl, 0, lo, kh);
    v16bf g1 = load_afrag_lds(hl, 1, lo, kh);
    #pragma unroll
    for (int nt = 0; nt < 4; ++nt) {
      v8f u = {};
      u = wmma_bf16(g0, load_bfrag(pk, FR_W2V + n * 8 + nt, lane), u);
      u = wmma_bf16(g1, load_bfrag(pk, FR_W2V + n * 8 + 4 + nt, lane), u);
      float bb = b2v[n * 64 + nt * 16 + lo];
      #pragma unroll
      for (int r = 0; r < 8; ++r)
        hl[(r + 8 * kh) * 64 + nt * 16 + lo] = (__bf16)(u[r] + bb);
    }
    g0 = load_afrag_lds(hl, 0, lo, kh);
    g1 = load_afrag_lds(hl, 1, lo, kh);
    v8f tv[4] = {}, tg[4] = {};
    #pragma unroll
    for (int nt = 0; nt < 4; ++nt) {
      tv[nt] = wmma_bf16(g0, load_bfrag(pk, FR_WVV + n * 8 + nt, lane), tv[nt]);
      tv[nt] = wmma_bf16(g1, load_bfrag(pk, FR_WVV + n * 8 + 4 + nt, lane), tv[nt]);
      tg[nt] = wmma_bf16(g0, load_bfrag(pk, FR_WGV + n * 8 + nt, lane), tg[nt]);
      tg[nt] = wmma_bf16(g1, load_bfrag(pk, FR_WGV + n * 8 + 4 + nt, lane), tg[nt]);
    }
    float y4[4][8];
    #pragma unroll
    for (int nt = 0; nt < 4; ++nt) {
      float bv_ = bvv[n * 64 + nt * 16 + lo];
      float bg_ = bgv[n * 64 + nt * 16 + lo];
      #pragma unroll
      for (int r = 0; r < 8; ++r) {
        float vv = tv[nt][r] + bv_;
        float gg = tg[nt][r] + bg_;
        float gl = vv / (1.f + __expf(-gg));
        float e = emb[(size_t)(row0 + r + 8 * kh) * FDIM + n * 64 + nt * 16 + lo];
        y4[nt][r] = e + gl;   // identity skip
      }
    }
    float gvp[4], bep[4];
    #pragma unroll
    for (int nt = 0; nt < 4; ++nt) {
      gvp[nt] = gv[n * 64 + nt * 16 + lo];
      bep[nt] = betav[n * 64 + nt * 16 + lo];
    }
    #pragma unroll
    for (int r = 0; r < 8; ++r) {
      float s = 0.f, q = 0.f;
      #pragma unroll
      for (int nt = 0; nt < 4; ++nt) { s += y4[nt][r]; q += y4[nt][r] * y4[nt][r]; }
      s = redsum16(s); q = redsum16(q);
      float mu = s * (1.f / 64.f);
      float var = q * (1.f / 64.f) - mu * mu;
      float rs = rsqrtf(var + EPSF);
      float w = __shfl(sw[r], n, 16);   // broadcast sparse weight for variable n
      #pragma unroll
      for (int nt = 0; nt < 4; ++nt) {
        float tval = gvp[nt] * (y4[nt][r] - mu) * rs + bep[nt];
        sv[nt][r] += w * tval;
      }
    }
  }
  #pragma unroll
  for (int nt = 0; nt < 4; ++nt)
    #pragma unroll
    for (int r = 0; r < 8; ++r)
      out_static[(size_t)(row0 + r + 8 * kh) * 64 + nt * 16 + lo] = sv[nt][r];
}

extern "C" void kernel_launch(void* const* d_in, const int* in_sizes, int n_in,
                              void* d_out, int out_size, void* d_ws, size_t ws_size,
                              hipStream_t stream) {
  const float* emb   = (const float*)d_in[0];
  const float* W1f   = (const float*)d_in[1];
  const float* b1f   = (const float*)d_in[2];
  const float* W2f   = (const float*)d_in[3];
  const float* b2f   = (const float*)d_in[4];
  const float* Wvf   = (const float*)d_in[5];
  const float* bvf   = (const float*)d_in[6];
  const float* Wgf   = (const float*)d_in[7];
  const float* bgf   = (const float*)d_in[8];
  const float* Wsf   = (const float*)d_in[9];
  const float* bsf   = (const float*)d_in[10];
  const float* gf    = (const float*)d_in[11];
  const float* betaf = (const float*)d_in[12];
  const float* W1v   = (const float*)d_in[13];
  const float* b1v   = (const float*)d_in[14];
  const float* W2v   = (const float*)d_in[15];
  const float* b2v   = (const float*)d_in[16];
  const float* Wvv   = (const float*)d_in[17];
  const float* bvv   = (const float*)d_in[18];
  const float* Wgv   = (const float*)d_in[19];
  const float* bgv   = (const float*)d_in[20];
  const float* gv    = (const float*)d_in[21];
  const float* betav = (const float*)d_in[22];

  u32* pk = (u32*)d_ws;  // needs FR_TOTAL * 1024 B ~= 700 KB
  int Bsz = in_sizes[0] / (NVAR * HDIM);

  pack_weights<<<FR_TOTAL, 256, 0, stream>>>(W1f, W2f, Wvf, Wgf, Wsf,
                                             W1v, W2v, Wvv, Wgv, pk);

  float* out_static = (float*)d_out;
  float* out_sw = out_static + (size_t)Bsz * HDIM;
  int tiles = (Bsz + 15) / 16;
  int blocks = (tiles + 7) / 8;
  vsn_fused<<<blocks, 256, 0, stream>>>(emb, b1f, b2f, bvf, bgf, bsf, gf, betaf,
                                        b1v, b2v, bvv, bgv, gv, betav, pk,
                                        out_static, out_sw, Bsz);
}
